// SelfTokenFoam_83794811945122
// MI455X (gfx1250) — matrix operand order repaired and checked by reading, not compiled, not run on hardware
//
#include <hip/hip_runtime.h>

// ---------------- CDNA5 WMMA types ----------------
typedef __attribute__((ext_vector_type(16))) __bf16 v16bf;
typedef __attribute__((ext_vector_type(8)))  float  v8f;

union BFrag { v16bf v; uint4 q[2]; };

#define Vv 32000
#define Dd 256
#define Kk 8
#define NBb 4
#define Tt 16

// ---------------- helpers ----------------
__device__ __forceinline__ float blockReduceSum(float v, float* red) {
    int tid = threadIdx.x;
    red[tid] = v; __syncthreads();
    for (int s = 128; s > 0; s >>= 1) {
        if (tid < s) red[tid] += red[tid + s];
        __syncthreads();
    }
    float r = red[0]; __syncthreads();
    return r;
}

// reduce 8 independent sums at once (one tree, 8 lanes of work per thread)
__device__ __forceinline__ void blockReduceSum8(const float v[8], float* out8, float* red) {
    int tid = threadIdx.x;
    for (int k = 0; k < 8; ++k) red[k * 256 + tid] = v[k];
    __syncthreads();
    for (int s = 128; s > 0; s >>= 1) {
        if (tid < s)
            for (int k = 0; k < 8; ++k) red[k * 256 + tid] += red[k * 256 + tid + s];
        __syncthreads();
    }
    if (tid < 8) out8[tid] = red[tid * 256];
    __syncthreads();
}

// ---------------- K1: fp32 -> bf16 convert ----------------
__global__ void cvt_bf16_kernel(const float* __restrict__ src, __bf16* __restrict__ dst, int n) {
    int i = blockIdx.x * blockDim.x + threadIdx.x;
    int stride = gridDim.x * blockDim.x;
    for (; i < n; i += stride) dst[i] = (__bf16)src[i];
}

// ---------------- K2: serial foam recurrence (one workgroup, memory state in LDS) ----
__global__ void foam_serial_kernel(const int* __restrict__ tokens,
                                   const float* __restrict__ embed,
                                   const float* __restrict__ bubbles,
                                   const float* __restrict__ scalars,
                                   const float* __restrict__ noise,
                                   const __bf16* __restrict__ Wbf,
                                   float* __restrict__ eq_all) {
    __shared__ float mem_s[32 * 256];   // memory state (K*NB, D)
    __shared__ float mm_s[8 * 256];     // mm, then reused as xwm
    __shared__ float red[8 * 256];      // reduction scratch
    __shared__ float r8a[8], r8b[8];

    int tid = threadIdx.x;
    for (int i = tid; i < 32 * 256; i += 256) mem_s[i] = 0.f;
    __syncthreads();

    float ngate      = 1.f / (1.f + expf(-scalars[0]));
    float decay_base = scalars[2];
    float sens       = fabsf(scalars[3]);

    for (int t = 0; t < Tt; ++t) {
        int tok  = tokens[t];
        float xv = embed[(size_t)tok * 256 + tid];

        float mmv[8], v8a[8], v8b[8];
        for (int k = 0; k < 8; ++k) {
            float s = 0.f;
            for (int b = 0; b < 4; ++b) s += mem_s[(k * 4 + b) * 256 + tid];
            mmv[k] = 0.25f * s;
            mm_s[k * 256 + tid] = mmv[k];
            v8a[k] = mmv[k] * mmv[k];
            v8b[k] = mmv[k] * xv;
        }
        blockReduceSum8(v8a, r8a, red);              // ||mm||^2 per foam
        blockReduceSum8(v8b, r8b, red);              // mm . x  per foam
        float xn2    = blockReduceSum(xv * xv, red);
        float x_norm = sqrtf(xn2) + 1e-10f;

        float decay[8], ssv[8];
        for (int k = 0; k < 8; ++k) {
            float mem_norm = sqrtf(r8a[k]) + 1e-10f;
            float nov = (mem_norm > 1e-8f) ? (1.f - r8b[k] / (x_norm * mem_norm)) : 1.f;
            decay[k] = 1.f / (1.f + expf(-(decay_base - sens * nov)));
            float xw = xv + decay[k] * mmv[k];
            mm_s[k * 256 + tid] = xw;                // mm buffer now holds xwm
            v8a[k] = xw * xw;
        }
        blockReduceSum8(v8a, r8a, red);              // ||xwm||^2 (syncs -> xwm visible)
        for (int k = 0; k < 8; ++k) ssv[k] = sqrtf(r8a[k]) + 1e-10f;

        // eq = tanh(xwm @ W_k) + bubbles + gated noise ; memory update
        for (int k = 0; k < 8; ++k) {
            float acc = 0.f;
            const __bf16* wk = Wbf + (size_t)k * 65536 + tid;
            #pragma unroll 4
            for (int d = 0; d < 256; ++d)
                acc += mm_s[k * 256 + d] * (float)wk[(size_t)d * 256];
            float th = tanhf(acc);
            float dk = decay[k];
            for (int b = 0; b < 4; ++b) {
                int idx = (k * 4 + b) * 256 + tid;
                float eqv = th + bubbles[idx] + ngate * noise[(size_t)t * 8192 + idx] * ssv[k] * 0.01f;
                mem_s[idx] = dk * mem_s[idx] + (1.f - dk) * eqv;
                eq_all[(size_t)t * 8192 + idx] = eqv;
            }
        }
        __syncthreads();
    }
}

// ---------------- K3: per-step postprocessing (parallel over t) --------------------
__global__ void postfoam_kernel(const float* __restrict__ eq_all,
                                const float* __restrict__ scalars,
                                __bf16* __restrict__ Abf,
                                float* __restrict__ rho_out,
                                float* __restrict__ S_out,
                                float* __restrict__ pur_out,
                                float* __restrict__ w_out) {
    int t = blockIdx.x, tid = threadIdx.x;
    __shared__ float eq_s[32 * 256];
    __shared__ float red[256];
    __shared__ float invn[32], scaleb[32], scale2_s[32];
    __shared__ float gram[32 * 32];
    __shared__ float Gp_s[32 * 32];
    __shared__ float w_s[8], pur_s[8];

    for (int i = tid; i < 8192; i += 256) eq_s[i] = eq_all[(size_t)t * 8192 + i];
    __syncthreads();

    for (int r = 0; r < 32; ++r) {
        float v = eq_s[r * 256 + tid];
        float s = blockReduceSum(v * v, red);
        if (tid == 0) invn[r] = 1.f / (sqrtf(s) + 1e-10f);
    }
    __syncthreads();

    // normalized Gram of the 32 rows
    for (int p = tid; p < 1024; p += 256) {
        int i = p >> 5, j = p & 31;
        float s = 0.f;
        for (int d = 0; d < 256; ++d) s += eq_s[i * 256 + d] * eq_s[j * 256 + d];
        gram[p] = s * invn[i] * invn[j];
    }
    __syncthreads();

    // purity_k = (1/NB^2) * sum_{b,b'} (m_b . m_b')^2
    if (tid < 8) {
        float s = 0.f;
        for (int b = 0; b < 4; ++b)
            for (int b2 = 0; b2 < 4; ++b2) {
                float g = gram[(tid * 4 + b) * 32 + (tid * 4 + b2)];
                s += g * g;
            }
        pur_s[tid] = s * (1.f / 16.f);
        pur_out[t * 8 + tid] = pur_s[tid];
    }
    __syncthreads();
    if (tid == 0) {
        float temp = fmaxf(fabsf(scalars[1]), 0.01f);
        float mx = -1e30f;
        for (int k = 0; k < 8; ++k) mx = fmaxf(mx, pur_s[k] / temp);
        float ssum = 0.f, e[8];
        for (int k = 0; k < 8; ++k) { e[k] = expf(pur_s[k] / temp - mx); ssum += e[k]; }
        for (int k = 0; k < 8; ++k) { w_s[k] = e[k] / ssum; w_out[t * 8 + k] = w_s[k]; }
    }
    __syncthreads();
    if (tid < 32) {
        float w  = w_s[tid >> 2];
        float sc = sqrtf(w * 0.25f) * invn[tid];
        scaleb[tid]   = sc;
        scale2_s[tid] = sc * sc;
    }
    __syncthreads();

    // A rows (bf16, row-major (j,d)) for the WMMA GEMM: rho_self = A^T A
    for (int i = tid; i < 8192; i += 256) {
        int r = i >> 8;
        Abf[(size_t)t * 8192 + i] = (__bf16)(eq_s[i] * scaleb[r]);
    }
    // weighted 32x32 Gram for eigenvalues of rho_self (rank-32)
    for (int p = tid; p < 1024; p += 256) {
        int i = p >> 5, j = p & 31;
        Gp_s[p] = 0.25f * sqrtf(w_s[i >> 2] * w_s[j >> 2]) * gram[p];
    }

    // rho_self output (fp32): rho[d][e] = sum_r scale2_r * eq[r][d]*eq[r][e]
    float sc2l[32];
    for (int r = 0; r < 32; ++r) sc2l[r] = scale2_s[r];
    for (int o = tid; o < 65536; o += 256) {
        int d = o >> 8, e2 = o & 255;
        float s = 0.f;
        #pragma unroll 8
        for (int r = 0; r < 32; ++r) s += sc2l[r] * eq_s[r * 256 + d] * eq_s[r * 256 + e2];
        rho_out[(size_t)t * 65536 + o] = s;
    }
    __syncthreads();

    // one-wave cyclic Jacobi on 32x32 (lockstep wave32, in-order LDS)
    if (tid < 32) {
        volatile float* A = Gp_s;
        int j = tid;
        for (int sweep = 0; sweep < 10; ++sweep)
            for (int p = 0; p < 31; ++p)
                for (int q = p + 1; q < 32; ++q) {
                    float apq = A[p * 32 + q];
                    if (fabsf(apq) > 1e-18f) {
                        float app = A[p * 32 + p], aqq = A[q * 32 + q];
                        float theta = 0.5f * (aqq - app) / apq;
                        float tt = ((theta >= 0.f) ? 1.f : -1.f) /
                                   (fabsf(theta) + sqrtf(theta * theta + 1.f));
                        float c = 1.f / sqrtf(tt * tt + 1.f), sn = tt * c;
                        float rp = c * A[p * 32 + j] - sn * A[q * 32 + j];
                        float rq = sn * A[p * 32 + j] + c * A[q * 32 + j];
                        __builtin_amdgcn_wave_barrier();
                        A[p * 32 + j] = rp; A[q * 32 + j] = rq;
                        __builtin_amdgcn_wave_barrier();
                        float cp = c * A[j * 32 + p] - sn * A[j * 32 + q];
                        float cq = sn * A[j * 32 + p] + c * A[j * 32 + q];
                        __builtin_amdgcn_wave_barrier();
                        A[j * 32 + p] = cp; A[j * 32 + q] = cq;
                        __builtin_amdgcn_wave_barrier();
                    }
                }
    }
    __syncthreads();
    if (tid == 0) {
        // 32 Gram eigenvalues + 224 clipped-at-1e-12 zeros of the rank-32 rho_self
        float sum = 224.0e-12f, lam[32];
        for (int i = 0; i < 32; ++i) {
            float l = fmaxf(Gp_s[i * 32 + i], 1e-12f);
            lam[i] = l; sum += l;
        }
        float S = 0.f;
        for (int i = 0; i < 32; ++i) {
            float p = lam[i] / sum;
            S -= p * fmaxf(logf(p), -100.f);
        }
        float pt = 1e-12f / sum;
        S -= 224.f * (pt * fmaxf(logf(pt), -100.f));
        S_out[t] = S;
    }
}

// ---------------- K4: logits = ||A e_v||^2 via bf16 WMMA (one 16x32 tile / wave) ----
__global__ void logits_wmma_kernel(const __bf16* __restrict__ Ebf,
                                   const __bf16* __restrict__ Abf,
                                   float* __restrict__ out) {
    int t    = blockIdx.y;
    int wave = threadIdx.x >> 5;
    int lane = threadIdx.x & 31;
    int g    = lane >> 4;          // lane half-group
    int l16  = lane & 15;
    int tile = blockIdx.x * 8 + wave;    // 0..1999
    int m0   = tile * 16;

    const __bf16* erow  = Ebf + (size_t)(m0 + l16) * 256;            // A-matrix row M=l16
    const __bf16* arow0 = Abf + (size_t)t * 8192 + (size_t)l16 * 256; // B cols j = l16
    const __bf16* arow1 = arow0 + 16 * 256;                           // B cols j = 16+l16

    v8f c0 = {}, c1 = {};
    #pragma unroll
    for (int c = 0; c < 8; ++c) {
        // A 16x32 bf16 frag: lane holds K in {8g..8g+7} U {16+8g..16+8g+7}
        BFrag a, b0, b1;
        a.q[0] = *(const uint4*)(erow + c * 32 + 8 * g);
        a.q[1] = *(const uint4*)(erow + c * 32 + 16 + 8 * g);
        // B 32x16 bf16 frag: lane holds K in {8g..8g+7} U {16+8g..16+8g+7}
        const __bf16* bp0 = arow0 + c * 32 + 8 * g;
        b0.q[0] = *(const uint4*)(bp0);
        b0.q[1] = *(const uint4*)(bp0 + 16);
        const __bf16* bp1 = arow1 + c * 32 + 8 * g;
        b1.q[0] = *(const uint4*)(bp1);
        b1.q[1] = *(const uint4*)(bp1 + 16);
        c0 = __builtin_amdgcn_wmma_f32_16x16x32_bf16(false, a.v, false, b0.v, (short)0, c0, false, false);
        c1 = __builtin_amdgcn_wmma_f32_16x16x32_bf16(false, a.v, false, b1.v, (short)0, c1, false, false);
    }

    // logits row m = sum_j P[m][j]^2 ; C layout: VGPR r -> rows r (lanes 0-15) / r+8 (16-31)
    float* lrow = out + (size_t)t * Vv + m0;
    #pragma unroll
    for (int r = 0; r < 8; ++r) {
        float v = c0[r] * c0[r] + c1[r] * c1[r];
        v += __shfl_xor(v, 1);
        v += __shfl_xor(v, 2);
        v += __shfl_xor(v, 4);
        v += __shfl_xor(v, 8);
        if (l16 == 0) lrow[r + 8 * g] = v;
    }
}

// ---------------- K5: in-place softmax over V + token entropy ----------------------
__global__ void softmax_h_kernel(float* __restrict__ probs, float* __restrict__ H_out) {
    int t = blockIdx.x, tid = threadIdx.x;
    __shared__ float red[256];
    float* L = probs + (size_t)t * Vv;

    float mx = -1e30f;
    for (int i = tid; i < Vv; i += 256) mx = fmaxf(mx, L[i]);
    red[tid] = mx; __syncthreads();
    for (int s = 128; s > 0; s >>= 1) { if (tid < s) red[tid] = fmaxf(red[tid], red[tid + s]); __syncthreads(); }
    mx = red[0]; __syncthreads();

    float ssum = 0.f;
    for (int i = tid; i < Vv; i += 256) ssum += expf(L[i] - mx);
    float Z = blockReduceSum(ssum, red);
    float logZ = mx + logf(Z);

    float hacc = 0.f;
    for (int i = tid; i < Vv; i += 256) {
        float lp = L[i] - logZ;
        float p  = expf(lp);
        L[i] = p;
        hacc += p * fmaxf(lp, -100.f);
    }
    float H = blockReduceSum(hacc, red);
    if (tid == 0) H_out[t] = -H;
}

// ---------------- launcher ----------------
extern "C" void kernel_launch(void* const* d_in, const int* in_sizes, int n_in,
                              void* d_out, int out_size, void* d_ws, size_t ws_size,
                              hipStream_t stream) {
    (void)in_sizes; (void)n_in; (void)out_size; (void)ws_size;

    const int*   tokens  = (const int*)d_in[0];
    const float* embed   = (const float*)d_in[1];
    const float* W       = (const float*)d_in[2];
    const float* bubbles = (const float*)d_in[3];
    const float* scalars = (const float*)d_in[4];
    const float* noise   = (const float*)d_in[5];

    float* out       = (float*)d_out;
    float* probs_out = out;                                   // (T,V)
    float* rho_out   = out + (size_t)Tt * Vv;                 // (T,D,D)
    float* S_out     = rho_out + (size_t)Tt * Dd * Dd;        // (T,)
    float* H_out     = S_out + Tt;                            // (T,)
    float* pur_out   = H_out + Tt;                            // (T,K)
    float* w_out     = pur_out + (size_t)Tt * Kk;             // (T,K)

    char* ws = (char*)d_ws;
    __bf16* Ebf   = (__bf16*)ws; ws += (size_t)Vv * Dd * 2;        // 16.38 MB
    __bf16* Wbf   = (__bf16*)ws; ws += (size_t)Kk * Dd * Dd * 2;   // 1.05 MB
    float*  eq_all= (float*)ws;  ws += (size_t)Tt * 32 * Dd * 4;   // 0.52 MB
    __bf16* Abf   = (__bf16*)ws;                                   // 0.26 MB

    cvt_bf16_kernel<<<512, 256, 0, stream>>>(embed, Ebf, Vv * Dd);
    cvt_bf16_kernel<<<256, 256, 0, stream>>>(W, Wbf, Kk * Dd * Dd);
    foam_serial_kernel<<<1, 256, 0, stream>>>(tokens, embed, bubbles, scalars, noise, Wbf, eq_all);
    postfoam_kernel<<<Tt, 256, 0, stream>>>(eq_all, scalars, Abf, rho_out, S_out, pur_out, w_out);
    logits_wmma_kernel<<<dim3(Vv / 128, Tt), 256, 0, stream>>>(Ebf, Abf, probs_out);
    softmax_h_kernel<<<Tt, 256, 0, stream>>>(probs_out, H_out);
}
